// _SolarOpenDecoderLayer_35691178230204
// MI455X (gfx1250) — matrix operand order repaired and checked by reading, not compiled, not run on hardware
//
#include <hip/hip_runtime.h>
#include <math.h>

// ---------------------------------------------------------------------------
// SolarOpen decoder layer for MI455X (gfx1250, wave32, WMMA).
// - All matmuls via v_wmma_f32_16x16x32_bf16 (f32 accum). bf16 weight set
//   (~61MB) is converted once per launch and stays resident in the 192MB L2.
// - MoE computes only the top-4 selected experts via expert-sorted grouped
//   GEMMs (identical math: combine weights are zero for unselected experts).
// - GEMM tiles stream global->LDS through GLOBAL_LOAD_ASYNC_TO_LDS_B128
//   (ASYNCcnt) with a double-buffered software pipeline; falls back to
//   synchronous staging if the builtin is unavailable.
// ---------------------------------------------------------------------------

typedef __attribute__((ext_vector_type(16))) __bf16 bf16x16;
typedef __attribute__((ext_vector_type(8)))  float  f32x8;
typedef int v4i_ __attribute__((vector_size(16)));   // matches async-LDS builtin

#define H_   1024
#define NH_  16
#define NKV_ 4
#define HD_  64
#define E_   16
#define I_   512
#define TK_  4
#define SEQ_ 1024
#define B_   2
#define T_   (B_ * SEQ_)

#if defined(__HIP_DEVICE_COMPILE__) && __has_builtin(__builtin_amdgcn_global_load_async_to_lds_b128)
#define HAVE_ASYNC_LDS 1
#else
#define HAVE_ASYNC_LDS 0
#endif

__device__ __forceinline__ unsigned short f2bf(float f) {
  union { float f; unsigned u; } x; x.f = f;
  unsigned r = x.u + 0x7FFFu + ((x.u >> 16) & 1u);
  return (unsigned short)(r >> 16);
}
__device__ __forceinline__ float bf2f(unsigned short h) {
  union { unsigned u; float f; } x; x.u = ((unsigned)h) << 16;
  return x.f;
}

// 16-byte global -> LDS copy: async (ASYNCcnt) on gfx1250, sync fallback.
__device__ __forceinline__ void copy16_to_lds(unsigned short* lds,
                                              const unsigned short* g) {
#if HAVE_ASYNC_LDS
  __builtin_amdgcn_global_load_async_to_lds_b128(
      (__attribute__((address_space(1))) v4i_*)g,
      (__attribute__((address_space(3))) v4i_*)lds, 0, 0);
#else
  *(uint4*)lds = *(const uint4*)g;
#endif
}

__device__ __forceinline__ void wait_async() {
#if HAVE_ASYNC_LDS
#if __has_builtin(__builtin_amdgcn_s_wait_asynccnt)
  __builtin_amdgcn_s_wait_asynccnt(0);
#else
  asm volatile("s_wait_asynccnt 0" ::: "memory");
#endif
#endif
}

union FragAB {
  struct { uint4 lo; uint4 hi; } u;
  bf16x16 v;
};

__device__ __forceinline__ f32x8 wmma_bf16(const FragAB& a, const FragAB& b, f32x8 c) {
  return __builtin_amdgcn_wmma_f32_16x16x32_bf16(false, a.v, false, b.v,
                                                 (short)0, c, false, false);
}

// --------------------------- elementwise helpers ---------------------------

__global__ void cast_bf16_kernel(const float* __restrict__ in,
                                 unsigned short* __restrict__ out, long n) {
  long i = (long)blockIdx.x * blockDim.x + threadIdx.x;
  long stride = (long)gridDim.x * blockDim.x;
  for (; i < n; i += stride) out[i] = f2bf(in[i]);
}

__global__ void rmsnorm_kernel(const float* __restrict__ x,
                               const float* __restrict__ w,
                               unsigned short* __restrict__ out) {
  __shared__ float red[256];
  int t = blockIdx.x;
  const float* xr = x + (long)t * H_;
  float s = 0.f;
  for (int c = threadIdx.x; c < H_; c += 256) { float v = xr[c]; s += v * v; }
  red[threadIdx.x] = s; __syncthreads();
  for (int off = 128; off > 0; off >>= 1) {
    if (threadIdx.x < off) red[threadIdx.x] += red[threadIdx.x + off];
    __syncthreads();
  }
  float rs = rsqrtf(red[0] / (float)H_ + 1e-6f);
  for (int c = threadIdx.x; c < H_; c += 256)
    out[(long)t * H_ + c] = f2bf(w[c] * xr[c] * rs);
}

__global__ void silu_mul_kernel(const float* __restrict__ g,
                                const float* __restrict__ u,
                                int ldg, int ldu,
                                unsigned short* __restrict__ out, int ldo,
                                long rows, int inner) {
  long i = (long)blockIdx.x * blockDim.x + threadIdx.x;
  long n = rows * (long)inner;
  long stride = (long)gridDim.x * blockDim.x;
  for (; i < n; i += stride) {
    long r = i / inner; int c = (int)(i - r * inner);
    float gv = g[r * ldg + c], uv = u[r * ldu + c];
    float s = gv / (1.f + __expf(-gv));
    out[r * ldo + c] = f2bf(s * uv);
  }
}

// --------------------------------- GEMM ------------------------------------
// C[M,N] = A[M,K](bf16) * W[N,K](bf16)^T, f32 accum via WMMA.
// Tile 128x64x32, 8 waves (4M x 2N), each wave = 2x2 16x16 WMMA tiles.
// Double-buffered LDS, async global->LDS staging (issue k+1 while computing k).
// Grouped (per-expert) mode, A-row gather, four epilogues:
//   EPI 0: store          EPI 1: store acc + addend
//   EPI 2: C += acc       EPI 3: atomicAdd(C[tok], acc * rowScale)  (scatter)

#define BM 128
#define BN 64
#define BK 32
#define LDSS 40   // padded LDS row stride (elems): 80B, 16B-aligned, conflict-free

template <int EPI>
__global__ __launch_bounds__(256) void gemm_bf16_kernel(
    const unsigned short* __restrict__ A,
    const unsigned short* __restrict__ W,
    float* __restrict__ C,
    int M, int N, int K, int lda, int ldc,
    const int* __restrict__ grpOff, int mtPerGrp, long wStride,
    const int* __restrict__ gather,
    const float* __restrict__ addend,
    const int* __restrict__ scatterTok,
    const float* __restrict__ rowScale) {
  __shared__ __align__(16) unsigned short As[2][BM * LDSS];
  __shared__ __align__(16) unsigned short Bs[2][BN * LDSS];
  int rowBase = 0, rows = M, mt = blockIdx.x;
  const unsigned short* Wp = W;
  if (grpOff) {
    int e = blockIdx.x / mtPerGrp; mt = blockIdx.x % mtPerGrp;
    rowBase = grpOff[e]; rows = grpOff[e + 1] - rowBase;
    Wp = W + (long)e * wStride;
  }
  int m0 = mt * BM;
  if (m0 >= rows) return;
  int n0 = blockIdx.y * BN;
  int tid = threadIdx.x;
  int wave = tid >> 5, lane = tid & 31;
  int wm = wave & 3, wn = wave >> 2;
  int lhalf = lane >> 4, lm = lane & 15;

  f32x8 acc[2][2] = {};

  auto stage = [&](int k0, int buf) {
    // A tile (gathered, zero-padded past group rows): 512 16B chunks
    for (int c = tid; c < (BM * BK / 8); c += 256) {
      int r = c >> 2, col = (c & 3) * 8;
      int gr = m0 + r;
      unsigned short* dst = &As[buf][r * LDSS + col];
      if (gr < rows) {
        long arow = gather ? (long)gather[rowBase + gr] : (long)(rowBase + gr);
        copy16_to_lds(dst, A + arow * lda + k0 + col);
      } else {
        *(uint4*)dst = make_uint4(0u, 0u, 0u, 0u);
      }
    }
    // B tile (weight rows are K-contiguous = WMMA B layout): 256 chunks
    int r = tid >> 2, col = (tid & 3) * 8;
    copy16_to_lds(&Bs[buf][r * LDSS + col], Wp + (long)(n0 + r) * K + k0 + col);
  };

  int nk = K / BK;
  stage(0, 0);
  for (int kk = 0; kk < nk; kk++) {
    int cur = kk & 1;
    wait_async();        // tile kk landed in LDS (this wave's asyncs)
    __syncthreads();     // ... and everyone else's
    if (kk + 1 < nk) stage(kk + 1, cur ^ 1);  // stream next tile during compute

    FragAB a[2], b[2];
    for (int i = 0; i < 2; i++) {
      // A frag: lane m holds row m; K pairs per VGPR (16b A 16x32 layout)
      const uint4* p = (const uint4*)(&As[cur][(wm * 32 + i * 16 + lm) * LDSS + lhalf * 8]);
      a[i].u.lo = p[0]; a[i].u.hi = p[2];
    }
    for (int j = 0; j < 2; j++) {
      // B frag: lane n holds column n; K pairs per VGPR (same k order as A)
      const uint4* p = (const uint4*)(&Bs[cur][(wn * 32 + j * 16 + lm) * LDSS + lhalf * 8]);
      b[j].u.lo = p[0]; b[j].u.hi = p[2];
    }
    for (int i = 0; i < 2; i++)
      for (int j = 0; j < 2; j++)
        acc[i][j] = wmma_bf16(a[i], b[j], acc[i][j]);
    __syncthreads();     // all waves done reading buffer `cur`
  }

  // epilogue: C/D layout -> lane holds column lm, rows r + 8*lhalf
  for (int i = 0; i < 2; i++) {
    int rbase = m0 + wm * 32 + i * 16 + lhalf * 8;
    for (int j = 0; j < 2; j++) {
      int col = n0 + wn * 32 + j * 16 + lm;
      for (int r = 0; r < 8; r++) {
        int rr = rbase + r;
        if (rr >= rows) continue;
        float v = acc[i][j][r];
        long crow = (long)rowBase + rr;
        if (EPI == 0) {
          C[crow * ldc + col] = v;
        } else if (EPI == 1) {
          C[crow * ldc + col] = v + addend[crow * ldc + col];
        } else if (EPI == 2) {
          C[crow * ldc + col] += v;
        } else {
          long tok = scatterTok[crow];
          atomicAdd(&C[tok * ldc + col], v * rowScale[crow]);
        }
      }
    }
  }
}

// --------------------------------- RoPE ------------------------------------

__global__ void rope_kernel(const float* __restrict__ qf,
                            const float* __restrict__ kf,
                            const float* __restrict__ vf,
                            const long long* __restrict__ pos_ids,
                            unsigned short* __restrict__ qb,
                            unsigned short* __restrict__ kb,
                            unsigned short* __restrict__ vb) {
  int t = blockIdx.x;
  float pos = (float)pos_ids[t];
  int tid = threadIdx.x;
  for (int idx = tid; idx < NH_ * HD_; idx += 256) {
    int d = idx & (HD_ - 1);
    int fi = d & 31;
    float inv = __powf(10000.f, -((float)fi) / 32.f);
    float sn, cs; __sincosf(pos * inv, &sn, &cs);
    long base = (long)t * (NH_ * HD_);
    float x = qf[base + idx];
    float other = (d < 32) ? -qf[base + idx + 32] : qf[base + idx - 32];
    qb[base + idx] = f2bf(x * cs + other * sn);
  }
  for (int idx = tid; idx < NKV_ * HD_; idx += 256) {
    int d = idx & (HD_ - 1);
    int fi = d & 31;
    float inv = __powf(10000.f, -((float)fi) / 32.f);
    float sn, cs; __sincosf(pos * inv, &sn, &cs);
    long base = (long)t * (NKV_ * HD_);
    float x = kf[base + idx];
    float other = (d < 32) ? -kf[base + idx + 32] : kf[base + idx - 32];
    kb[base + idx] = f2bf(x * cs + other * sn);
    vb[base + idx] = f2bf(vf[base + idx]);
  }
}

// ----------------------------- flash attention -----------------------------
// Block = (b,h, 128 q rows); 8 waves, one 16-row q tile per wave.
// QK^T and P*V both via WMMA; online softmax with width-16 shfl reductions
// matching the C-matrix lane layout (lane holds column lm; rows r + 8*half).
// K tile streams async to LDS; V is transposed during staging (ds stores).

__global__ __launch_bounds__(256) void flash_attn_kernel(
    const unsigned short* __restrict__ qb,   // [T][NH*HD]
    const unsigned short* __restrict__ kb,   // [T][NKV*HD]
    const unsigned short* __restrict__ vb,   // [T][NKV*HD]
    unsigned short* __restrict__ ob) {       // [T][NH*HD]
  __shared__ __align__(16) unsigned short Ks[32 * 72];   // [key][d], padded
  __shared__ __align__(16) unsigned short Vt[HD_ * 40];  // [d][key], padded
  __shared__ __align__(16) unsigned short Ps[8 * 16 * 40];
  int bh = blockIdx.x;
  int b = bh / NH_, h = bh % NH_;
  int hk = h / (NH_ / NKV_);
  int qblk = blockIdx.y;
  int tid = threadIdx.x, wave = tid >> 5, lane = tid & 31;
  int lhalf = lane >> 4, lm = lane & 15;
  int qrow0 = qblk * 128 + wave * 16;
  long qtok0 = (long)b * SEQ_ + qrow0;

  FragAB qa[2];
  {
    long base = (qtok0 + lm) * (NH_ * HD_) + h * HD_;
    for (int c2 = 0; c2 < 2; c2++) {
      const uint4* p = (const uint4*)(qb + base + c2 * 32 + lhalf * 8);
      qa[c2].u.lo = p[0]; qa[c2].u.hi = p[2];
    }
  }
  f32x8 oacc[4] = {};
  float mrun[8], lrun[8];
  for (int r = 0; r < 8; r++) { mrun[r] = -1e30f; lrun[r] = 0.f; }
  const float scale = 0.125f;  // 1/sqrt(64)
  int kvEnd = qblk * 128 + 128;

  for (int kt = 0; kt < kvEnd; kt += 32) {
    {  // stage K [key][d] (async) and V transposed [d][key]
      int r = tid >> 3, ch = tid & 7;
      long ktok = (long)b * SEQ_ + kt + r;
      copy16_to_lds(&Ks[r * 72 + ch * 8],
                    kb + ktok * (NKV_ * HD_) + hk * HD_ + ch * 8);
      uint4 vv4 = *(const uint4*)(vb + ktok * (NKV_ * HD_) + hk * HD_ + ch * 8);
      const unsigned short* ve = (const unsigned short*)&vv4;
      for (int i2 = 0; i2 < 8; i2++)
        Vt[(ch * 8 + i2) * 40 + r] = ve[i2];
    }
    wait_async();
    __syncthreads();
    if (kt <= qrow0 + 15) {
      f32x8 sacc[2] = {};
      for (int j = 0; j < 2; j++) {
        for (int c2 = 0; c2 < 2; c2++) {
          FragAB bfr;
          const uint4* p = (const uint4*)(&Ks[(j * 16 + lm) * 72 + c2 * 32 + lhalf * 8]);
          bfr.u.lo = p[0]; bfr.u.hi = p[2];
          sacc[j] = wmma_bf16(qa[c2], bfr, sacc[j]);
        }
      }
      float fac[8], p0[8], p1[8];
      for (int r = 0; r < 8; r++) {
        int qr = qrow0 + r + 8 * lhalf;
        float s0 = sacc[0][r] * scale;
        float s1 = sacc[1][r] * scale;
        if (kt + lm > qr) s0 = -1e30f;
        if (kt + 16 + lm > qr) s1 = -1e30f;
        float mx = fmaxf(s0, s1);
        for (int off = 1; off < 16; off <<= 1)
          mx = fmaxf(mx, __shfl_xor(mx, off, 16));
        float mnew = fmaxf(mrun[r], mx);
        fac[r] = __expf(mrun[r] - mnew);
        p0[r] = __expf(s0 - mnew);
        p1[r] = __expf(s1 - mnew);
        lrun[r] = lrun[r] * fac[r] + p0[r] + p1[r];
        mrun[r] = mnew;
      }
      for (int dt = 0; dt < 4; dt++)
        for (int r = 0; r < 8; r++)
          oacc[dt][r] *= fac[r];
      // P (C layout) -> LDS -> A fragment, then P*V via WMMA
      unsigned short* Pw = &Ps[wave * 16 * 40];
      for (int r = 0; r < 8; r++) {
        int m = r + 8 * lhalf;
        Pw[m * 40 + lm] = f2bf(p0[r]);
        Pw[m * 40 + 16 + lm] = f2bf(p1[r]);
      }
      FragAB pa;
      const uint4* pp = (const uint4*)(&Ps[(wave * 16 + lm) * 40 + lhalf * 8]);
      pa.u.lo = pp[0]; pa.u.hi = pp[2];
      for (int dt = 0; dt < 4; dt++) {
        FragAB vbfr;
        const uint4* vp = (const uint4*)(&Vt[(dt * 16 + lm) * 40 + lhalf * 8]);
        vbfr.u.lo = vp[0]; vbfr.u.hi = vp[2];
        oacc[dt] = wmma_bf16(pa, vbfr, oacc[dt]);
      }
    }
    __syncthreads();
  }
  float lsum[8];
  for (int r = 0; r < 8; r++) {
    float l = lrun[r];
    for (int off = 1; off < 16; off <<= 1)
      l += __shfl_xor(l, off, 16);
    lsum[r] = 1.f / l;
  }
  for (int dt = 0; dt < 4; dt++)
    for (int r = 0; r < 8; r++) {
      long tok = qtok0 + r + 8 * lhalf;
      ob[tok * (NH_ * HD_) + h * HD_ + dt * 16 + lm] = f2bf(oacc[dt][r] * lsum[r]);
    }
}

// ------------------------------ MoE routing --------------------------------

__global__ void routing_kernel(const unsigned short* __restrict__ xb,
                               const float* __restrict__ gate_w,
                               const float* __restrict__ e_bias,
                               int* __restrict__ topk_idx,
                               float* __restrict__ topk_w,
                               int* __restrict__ counts) {
  int wave = threadIdx.x >> 5, lane = threadIdx.x & 31;
  int t = blockIdx.x * 8 + wave;
  float logit = 0.f;
  if (lane < E_) {
    const unsigned short* xr = xb + (long)t * H_;
    const float* gw = gate_w + (long)lane * H_;
    float a = 0.f;
    for (int hh = 0; hh < H_; hh++) a += bf2f(xr[hh]) * gw[hh];
    logit = a;
  }
  float lg[E_];
  for (int e = 0; e < E_; e++) lg[e] = __shfl(logit, e, 32);
  if (lane == 0) {
    float sc[E_], sfc[E_];
    for (int e = 0; e < E_; e++) {
      sc[e] = 1.f / (1.f + __expf(-lg[e]));
      sfc[e] = sc[e] + e_bias[e];
    }
    float gsc[4];
    for (int g = 0; g < 4; g++) {
      float m = sfc[g * 4];
      for (int j = 1; j < 4; j++) m = fmaxf(m, sfc[g * 4 + j]);
      gsc[g] = m;
    }
    int g1 = 0;
    for (int g = 1; g < 4; g++) if (gsc[g] > gsc[g1]) g1 = g;
    int g2 = (g1 == 0) ? 1 : 0;
    for (int g = 0; g < 4; g++) if (g != g1 && gsc[g] > gsc[g2]) g2 = g;
    bool taken[E_] = {};
    int idxs[TK_];
    float wsum = 0.f;
    for (int k = 0; k < TK_; k++) {
      int best = 0; float bv = -1e30f;
      for (int e = 0; e < E_; e++) {
        int g = e >> 2;
        float v = (g == g1 || g == g2) ? sfc[e] : 0.f;  // masked -> 0 (ref)
        if (!taken[e] && v > bv) { bv = v; best = e; }
      }
      taken[best] = true; idxs[k] = best; wsum += sc[best];
    }
    float inv = 2.5f / (wsum + 1e-20f);
    for (int k = 0; k < TK_; k++) {
      topk_idx[t * TK_ + k] = idxs[k];
      topk_w[t * TK_ + k] = sc[idxs[k]] * inv;
      atomicAdd(&counts[idxs[k]], 1);
    }
  }
}

__global__ void scan_kernel(const int* __restrict__ counts,
                            int* __restrict__ offsets,
                            int* __restrict__ cursor) {
  if (threadIdx.x == 0 && blockIdx.x == 0) {
    int a = 0;
    for (int e = 0; e < E_; e++) { offsets[e] = a; cursor[e] = a; a += counts[e]; }
    offsets[E_] = a;
  }
}

__global__ void scatter_kernel(const int* __restrict__ topk_idx,
                               const float* __restrict__ topk_w,
                               int* __restrict__ cursor,
                               int* __restrict__ perm,
                               float* __restrict__ pw) {
  int t = blockIdx.x * blockDim.x + threadIdx.x;
  if (t >= T_) return;
  for (int k = 0; k < TK_; k++) {
    int e = topk_idx[t * TK_ + k];
    int slot = atomicAdd(&cursor[e], 1);
    perm[slot] = t;
    pw[slot] = topk_w[t * TK_ + k];
  }
}

// --------------------------------- launch ----------------------------------

extern "C" void kernel_launch(void* const* d_in, const int* in_sizes, int n_in,
                              void* d_out, int out_size, void* d_ws, size_t ws_size,
                              hipStream_t stream) {
  const float*     hidden = (const float*)d_in[0];
  const long long* pos    = (const long long*)d_in[1];
  const float*     ln1    = (const float*)d_in[2];
  const float*     ln2    = (const float*)d_in[3];
  const float*     q_w    = (const float*)d_in[4];
  const float*     k_w    = (const float*)d_in[5];
  const float*     v_w    = (const float*)d_in[6];
  const float*     o_w    = (const float*)d_in[7];
  const float*     gate_w = (const float*)d_in[8];
  const float*     e_bias = (const float*)d_in[9];
  const float*     W_gu   = (const float*)d_in[10];
  const float*     W_d    = (const float*)d_in[11];
  const float*     sg_w   = (const float*)d_in[12];
  const float*     su_w   = (const float*)d_in[13];
  const float*     sd_w   = (const float*)d_in[14];
  float* out = (float*)d_out;

  char* p = (char*)d_ws;
  auto alloc = [&](size_t bytes) -> char* {
    char* r = p; p += (bytes + 255) & ~(size_t)255; return r;
  };
  // bf16 weights (stay resident in 192MB L2 across the launch)
  unsigned short* wq  = (unsigned short*)alloc((size_t)1024 * 1024 * 2);
  unsigned short* wk  = (unsigned short*)alloc((size_t)256 * 1024 * 2);
  unsigned short* wv  = (unsigned short*)alloc((size_t)256 * 1024 * 2);
  unsigned short* wo  = (unsigned short*)alloc((size_t)1024 * 1024 * 2);
  unsigned short* wgu = (unsigned short*)alloc((size_t)E_ * 1024 * 1024 * 2);
  unsigned short* wd  = (unsigned short*)alloc((size_t)E_ * 1024 * 512 * 2);
  unsigned short* wsg = (unsigned short*)alloc((size_t)1024 * 1024 * 2);
  unsigned short* wsu = (unsigned short*)alloc((size_t)1024 * 1024 * 2);
  unsigned short* wsd = (unsigned short*)alloc((size_t)1024 * 1024 * 2);
  // activations
  unsigned short* xb1    = (unsigned short*)alloc((size_t)T_ * H_ * 2);
  float*          qf     = (float*)alloc((size_t)T_ * 1024 * 4);
  float*          kf     = (float*)alloc((size_t)T_ * 256 * 4);
  float*          vf     = (float*)alloc((size_t)T_ * 256 * 4);
  unsigned short* qbv    = (unsigned short*)alloc((size_t)T_ * 1024 * 2);
  unsigned short* kbv    = (unsigned short*)alloc((size_t)T_ * 256 * 2);
  unsigned short* vbv    = (unsigned short*)alloc((size_t)T_ * 256 * 2);
  unsigned short* attn_b = (unsigned short*)alloc((size_t)T_ * 1024 * 2);
  unsigned short* xb2    = (unsigned short*)alloc((size_t)T_ * H_ * 2);
  int*            tidx   = (int*)alloc((size_t)T_ * TK_ * 4);
  float*          tw     = (float*)alloc((size_t)T_ * TK_ * 4);
  int*            counts = (int*)alloc(64);
  int*            offs   = (int*)alloc(128);
  int*            cursor = (int*)alloc(64);
  int*            perm   = (int*)alloc((size_t)T_ * TK_ * 4);
  float*          pw     = (float*)alloc((size_t)T_ * TK_ * 4);
  float*          gu     = (float*)alloc((size_t)T_ * TK_ * 1024 * 4);
  unsigned short* hidb   = (unsigned short*)alloc((size_t)T_ * TK_ * 512 * 2);
  float*          suf    = (float*)alloc((size_t)T_ * 1024 * 4);
  float*          sgf    = qf;      // reuse (q consumed after RoPE)
  unsigned short* shb    = attn_b;  // reuse (attn consumed after o-proj)

  (void)hipMemsetAsync(counts, 0, 64, stream);

  // weight conversion f32 -> bf16
  auto cast = [&](const float* src, unsigned short* dst, long n) {
    cast_bf16_kernel<<<2048, 256, 0, stream>>>(src, dst, n);
  };
  cast(q_w, wq, 1024L * 1024); cast(k_w, wk, 256L * 1024); cast(v_w, wv, 256L * 1024);
  cast(o_w, wo, 1024L * 1024);
  cast(W_gu, wgu, (long)E_ * 1024 * 1024);
  cast(W_d, wd, (long)E_ * 1024 * 512);
  cast(sg_w, wsg, 1024L * 1024); cast(su_w, wsu, 1024L * 1024);
  cast(sd_w, wsd, 1024L * 1024);

  // ---- attention ----
  rmsnorm_kernel<<<T_, 256, 0, stream>>>(hidden, ln1, xb1);
  gemm_bf16_kernel<0><<<dim3(16, 16), 256, 0, stream>>>(
      xb1, wq, qf, T_, 1024, 1024, 1024, 1024,
      nullptr, 0, 0, nullptr, nullptr, nullptr, nullptr);
  gemm_bf16_kernel<0><<<dim3(16, 4), 256, 0, stream>>>(
      xb1, wk, kf, T_, 256, 1024, 1024, 256,
      nullptr, 0, 0, nullptr, nullptr, nullptr, nullptr);
  gemm_bf16_kernel<0><<<dim3(16, 4), 256, 0, stream>>>(
      xb1, wv, vf, T_, 256, 1024, 1024, 256,
      nullptr, 0, 0, nullptr, nullptr, nullptr, nullptr);
  rope_kernel<<<T_, 256, 0, stream>>>(qf, kf, vf, pos, qbv, kbv, vbv);
  flash_attn_kernel<<<dim3(B_ * NH_, SEQ_ / 128), 256, 0, stream>>>(qbv, kbv, vbv, attn_b);
  gemm_bf16_kernel<1><<<dim3(16, 16), 256, 0, stream>>>(
      attn_b, wo, out, T_, 1024, 1024, 1024, 1024,
      nullptr, 0, 0, nullptr, hidden, nullptr, nullptr);   // out = residual + attn

  // ---- MoE ----
  rmsnorm_kernel<<<T_, 256, 0, stream>>>(out, ln2, xb2);
  routing_kernel<<<T_ / 8, 256, 0, stream>>>(xb2, gate_w, e_bias, tidx, tw, counts);
  scan_kernel<<<1, 32, 0, stream>>>(counts, offs, cursor);
  scatter_kernel<<<(T_ + 255) / 256, 256, 0, stream>>>(tidx, tw, cursor, perm, pw);

  // routed experts: gather-GEMM -> silu*up -> scatter-atomic GEMM
  gemm_bf16_kernel<0><<<dim3(E_ * 16, 16), 256, 0, stream>>>(
      xb2, wgu, gu, 0, 1024, 1024, 1024, 1024,
      offs, 16, 1024L * 1024, perm, nullptr, nullptr, nullptr);
  silu_mul_kernel<<<4096, 256, 0, stream>>>(gu, gu + 512, 1024, 1024,
                                            hidb, 512, (long)T_ * TK_, 512);
  // shared expert
  gemm_bf16_kernel<0><<<dim3(16, 16), 256, 0, stream>>>(
      xb2, wsg, sgf, T_, 1024, 1024, 1024, 1024,
      nullptr, 0, 0, nullptr, nullptr, nullptr, nullptr);
  gemm_bf16_kernel<0><<<dim3(16, 16), 256, 0, stream>>>(
      xb2, wsu, suf, T_, 1024, 1024, 1024, 1024,
      nullptr, 0, 0, nullptr, nullptr, nullptr, nullptr);
  silu_mul_kernel<<<4096, 256, 0, stream>>>(sgf, suf, 1024, 1024,
                                            shb, 1024, (long)T_, 1024);
  gemm_bf16_kernel<2><<<dim3(16, 16), 256, 0, stream>>>(
      shb, wsd, out, T_, 1024, 1024, 1024, 1024,
      nullptr, 0, 0, nullptr, nullptr, nullptr, nullptr);  // out += shared
  gemm_bf16_kernel<3><<<dim3(E_ * 16, 16), 256, 0, stream>>>(
      hidb, wd, out, 0, 1024, 512, 512, 1024,
      offs, 16, 1024L * 512, nullptr, nullptr, perm, pw);  // out += routed
}
